// GAT_71373766524938
// MI455X (gfx1250) — matrix operand order repaired and checked by reading, not compiled, not run on hardware
//
#include <hip/hip_runtime.h>
#include <hip/hip_bf16.h>
#include <stdint.h>

typedef __bf16 bf16_t;
typedef __attribute__((ext_vector_type(16))) bf16_t v16bf;
typedef __attribute__((ext_vector_type(8)))  float  v8f;

#define CDIM 128
#define NEG_SLOPE 0.2f

// ---------- order-preserving float <-> uint encoding for atomic max ----------
__device__ __forceinline__ unsigned fenc(float f) {
    unsigned u = __float_as_uint(f);
    return (u & 0x80000000u) ? ~u : (u | 0x80000000u);
}
__device__ __forceinline__ float fdec(unsigned e) {
    unsigned u = (e & 0x80000000u) ? (e & 0x7fffffffu) : ~e;
    return __uint_as_float(u);
}
__device__ __forceinline__ float lrelu(float a) { return a > 0.f ? a : NEG_SLOPE * a; }

// ---------- f32 -> bf16 hi/lo split ----------
__global__ void k_cvt_bf16(const float* __restrict__ src, bf16_t* __restrict__ hi,
                           bf16_t* __restrict__ lo, int n) {
    int i = blockIdx.x * blockDim.x + threadIdx.x;
    if (i >= n) return;
    float f = src[i];
    bf16_t h = (bf16_t)f;
    hi[i] = h;
    lo[i] = (bf16_t)(f - (float)h);
}

// ---------- pack W [128,128] into per-lane B-fragment order (hi/lo) ----------
// entry idx = ((nt*4 + kc)*32 + lane)*8 + j2 ; value = pack(B[kb,co], B[kb+1,co])
// with B[k,co] = W[co*128 + k], kb = kc*32 + (lane<16?0:16) + 2*j2, co = nt*16 + (lane&15)
__global__ void k_pack_w(const float* __restrict__ W, unsigned* __restrict__ pkH,
                         unsigned* __restrict__ pkL) {
    int idx = blockIdx.x * blockDim.x + threadIdx.x;
    if (idx >= 8 * 4 * 32 * 8) return;
    int j2 = idx & 7;
    int lane = (idx >> 3) & 31;
    int kc = (idx >> 8) & 3;
    int nt = idx >> 10;
    int co = nt * 16 + (lane & 15);
    int kb = kc * 32 + ((lane < 16) ? 0 : 16) + 2 * j2;
    float w0 = W[co * CDIM + kb];
    float w1 = W[co * CDIM + kb + 1];
    bf16_t h0 = (bf16_t)w0, h1 = (bf16_t)w1;
    bf16_t l0 = (bf16_t)(w0 - (float)h0), l1 = (bf16_t)(w1 - (float)h1);
    union { bf16_t b[2]; unsigned u; } ph, pl;
    ph.b[0] = h0; ph.b[1] = h1;
    pl.b[0] = l0; pl.b[1] = l1;
    pkH[idx] = ph.u;
    pkL[idx] = pl.u;
}

// ---------- WMMA GEMM: C[N,128] = A[N,128] @ W^T, bf16 hi/lo split ----------
__global__ void k_gemm_wmma(const bf16_t* __restrict__ Ah, const bf16_t* __restrict__ Al,
                            const unsigned* __restrict__ BpkH, const unsigned* __restrict__ BpkL,
                            float* __restrict__ Cmat, int N) {
    const int lane = threadIdx.x & 31;
    const int wave = (blockIdx.x * blockDim.x + threadIdx.x) >> 5;
    const int tiles_m = (N + 15) >> 4;
    if (wave >= tiles_m * 8) return;
    const int mt = wave >> 3;
    const int nt = wave & 7;
    const int m0 = mt * 16;
    int row = m0 + (lane & 15);
    if (row >= N) row = N - 1;
    const int kbaseSel = (lane < 16) ? 0 : 8;  // A-frag K base in bf16 pairs: k = 16h + s, same order as packed B
    v8f acc = {};
    for (int kc = 0; kc < 4; ++kc) {
        const int kk = kc * 32;
        union { v16bf v; unsigned u[8]; } ah, al, bh, bl;
#pragma unroll
        for (int j2 = 0; j2 < 8; ++j2) {
            int K = kk + 2 * kbaseSel + 2 * j2;
            ah.u[j2] = *(const unsigned*)(Ah + (size_t)row * CDIM + K);
            al.u[j2] = *(const unsigned*)(Al + (size_t)row * CDIM + K);
        }
        const unsigned* bpH = BpkH + (((nt * 4 + kc) * 32 + lane) * 8);
        const unsigned* bpL = BpkL + (((nt * 4 + kc) * 32 + lane) * 8);
#pragma unroll
        for (int j2 = 0; j2 < 8; ++j2) { bh.u[j2] = bpH[j2]; bl.u[j2] = bpL[j2]; }
        acc = __builtin_amdgcn_wmma_f32_16x16x32_bf16(false, ah.v, false, bh.v, (short)0, acc, false, false);
        acc = __builtin_amdgcn_wmma_f32_16x16x32_bf16(false, ah.v, false, bl.v, (short)0, acc, false, false);
        acc = __builtin_amdgcn_wmma_f32_16x16x32_bf16(false, al.v, false, bh.v, (short)0, acc, false, false);
    }
    const int ncol = nt * 16 + (lane & 15);
    const int mbase = m0 + ((lane < 16) ? 0 : 8);
#pragma unroll
    for (int r = 0; r < 8; ++r) {
        int M = mbase + r;
        if (M < N) Cmat[(size_t)M * CDIM + ncol] = acc[r];
    }
}

// ---------- per-node attention dots: alpha_src/alpha_dst (wave per node) ----------
__global__ void k_node_alpha(const float* __restrict__ XL, const float* __restrict__ aS,
                             const float* __restrict__ aD, float* __restrict__ oS,
                             float* __restrict__ oD, int N) {
    int lane = threadIdx.x & 31;
    int node = (blockIdx.x * blockDim.x + threadIdx.x) >> 5;
    if (node >= N) return;
    float s = 0.f, d = 0.f;
#pragma unroll
    for (int j = 0; j < 4; ++j) {
        int c = lane + j * 32;
        float v = XL[(size_t)node * CDIM + c];
        s += v * aS[c];
        d += v * aD[c];
    }
#pragma unroll
    for (int off = 16; off; off >>= 1) {
        s += __shfl_xor(s, off, 32);
        d += __shfl_xor(d, off, 32);
    }
    if (lane == 0) { oS[node] = s; oD[node] = d; }
}

// ---------- v_e = We^T @ a_edge (collapses the full edge projection GEMM) ----------
__global__ void k_we_dot(const float* __restrict__ We, const float* __restrict__ aE,
                         float* __restrict__ vE, int ED) {
    int k = threadIdx.x;
    if (k >= ED) return;
    float s = 0.f;
    for (int c = 0; c < CDIM; ++c) s += We[c * ED + k] * aE[c];
    vE[k] = s;
}

__global__ void k_count(const long long* __restrict__ dst, float* __restrict__ cnt, int E) {
    int e = blockIdx.x * blockDim.x + threadIdx.x;
    if (e >= E) return;
    atomicAdd(&cnt[(int)dst[e]], 1.f);
}

// t_e = ea_e . v_e ; also accumulate per-dst sums for the mean self-loop attr
__global__ void k_edge_t(const float* __restrict__ ea, const float* __restrict__ vE,
                         const long long* __restrict__ dst, float* __restrict__ eBuf,
                         float* __restrict__ sumT, int E, int ED) {
    int e = blockIdx.x * blockDim.x + threadIdx.x;
    if (e >= E) return;
    float t = 0.f;
    for (int k = 0; k < ED; ++k) t += ea[(size_t)e * ED + k] * vE[k];
    eBuf[e] = t;
    atomicAdd(&sumT[(int)dst[e]], t);
}

// self-loop alpha; seeds the segment max (every node has a self loop)
__global__ void k_loop_init(const float* __restrict__ oS, const float* __restrict__ oD,
                            const float* __restrict__ sumT, const float* __restrict__ cnt,
                            float* __restrict__ pLoop, unsigned* __restrict__ mEnc, int N) {
    int n = blockIdx.x * blockDim.x + threadIdx.x;
    if (n >= N) return;
    float al = oS[n] + oD[n] + sumT[n] / fmaxf(cnt[n], 1.f);
    al = lrelu(al);
    pLoop[n] = al;
    mEnc[n] = fenc(al);
}

__global__ void k_edge_max(const long long* __restrict__ src, const long long* __restrict__ dst,
                           const float* __restrict__ oS, const float* __restrict__ oD,
                           float* __restrict__ eBuf, unsigned* __restrict__ mEnc, int E) {
    int e = blockIdx.x * blockDim.x + threadIdx.x;
    if (e >= E) return;
    int s = (int)src[e], d = (int)dst[e];
    float a = lrelu(oS[s] + oD[d] + eBuf[e]);
    eBuf[e] = a;
    atomicMax(&mEnc[d], fenc(a));
}

__global__ void k_node_exp(float* __restrict__ pLoop, const unsigned* __restrict__ mEnc,
                           float* __restrict__ sArr, int N) {
    int n = blockIdx.x * blockDim.x + threadIdx.x;
    if (n >= N) return;
    float p = expf(pLoop[n] - fdec(mEnc[n]));
    pLoop[n] = p;
    sArr[n] = p;  // seeds segment sum with self-loop term
}

__global__ void k_edge_exp(const long long* __restrict__ dst, float* __restrict__ eBuf,
                           const unsigned* __restrict__ mEnc, float* __restrict__ sArr, int E) {
    int e = blockIdx.x * blockDim.x + threadIdx.x;
    if (e >= E) return;
    int d = (int)dst[e];
    float p = expf(eBuf[e] - fdec(mEnc[d]));
    eBuf[e] = p;
    atomicAdd(&sArr[d], p);
}

// agg[n,:] = coef_self * x_l[n,:]   (full overwrite -> no memset needed)
__global__ void k_self_agg(const float* __restrict__ XL, const float* __restrict__ pLoop,
                           const float* __restrict__ sArr, float* __restrict__ agg, int N) {
    int lane = threadIdx.x & 31;
    int n = (blockIdx.x * blockDim.x + threadIdx.x) >> 5;
    if (n >= N) return;
    float coef = pLoop[n] / sArr[n];
    float4 v = ((const float4*)(XL + (size_t)n * CDIM))[lane];
    ((float4*)(agg + (size_t)n * CDIM))[lane] = make_float4(coef * v.x, coef * v.y, coef * v.z, coef * v.w);
}

// heavy kernel: wave per edge, gather x_l[src] + scatter-add into agg[dst] (L2-resident)
__global__ void k_edge_agg(const long long* __restrict__ src, const long long* __restrict__ dst,
                           const float* __restrict__ XL, const float* __restrict__ eBuf,
                           const float* __restrict__ sArr, float* __restrict__ agg, int E) {
    int lane = threadIdx.x & 31;
    int e = (blockIdx.x * blockDim.x + threadIdx.x) >> 5;
    if (e >= E) return;
    int s = (int)src[e], d = (int)dst[e];
    float coef = eBuf[e] / sArr[d];
    float4 v = ((const float4*)(XL + (size_t)s * CDIM))[lane];
    float* ap = agg + (size_t)d * CDIM + lane * 4;
    atomicAdd(ap + 0, coef * v.x);
    atomicAdd(ap + 1, coef * v.y);
    atomicAdd(ap + 2, coef * v.z);
    atomicAdd(ap + 3, coef * v.w);
}

__global__ void k_bias_silu(float* __restrict__ agg, const float* __restrict__ b, int total) {
    int i = blockIdx.x * blockDim.x + threadIdx.x;
    if (i >= total) return;
    float v = agg[i] + b[i & (CDIM - 1)];
    agg[i] = v / (1.f + expf(-v));
}

__global__ void k_gpool(const float* __restrict__ h, const long long* __restrict__ bid,
                        float* __restrict__ gsum, float* __restrict__ gcnt, int N) {
    int lane = threadIdx.x & 31;
    int n = (blockIdx.x * blockDim.x + threadIdx.x) >> 5;
    if (n >= N) return;
    int g = (int)bid[n];
    float4 v = ((const float4*)(h + (size_t)n * CDIM))[lane];
    float* gp = gsum + (size_t)g * CDIM + lane * 4;
    atomicAdd(gp + 0, v.x);
    atomicAdd(gp + 1, v.y);
    atomicAdd(gp + 2, v.z);
    atomicAdd(gp + 3, v.w);
    if (lane == 0) atomicAdd(&gcnt[g], 1.f);
}

__global__ void k_gdiv(const float* __restrict__ gsum, const float* __restrict__ gcnt,
                       float* __restrict__ out, int total) {
    int i = blockIdx.x * blockDim.x + threadIdx.x;
    if (i >= total) return;
    out[i] = gsum[i] / fmaxf(gcnt[i >> 7], 1.f);
}

extern "C" void kernel_launch(void* const* d_in, const int* in_sizes, int n_in,
                              void* d_out, int out_size, void* d_ws, size_t ws_size,
                              hipStream_t stream) {
    const float* x = (const float*)d_in[0];
    const long long* ei = (const long long*)d_in[1];
    const float* ea = (const float*)d_in[2];
    const long long* bid = (const long long*)d_in[3];
    const float* W[2]  = {(const float*)d_in[5],  (const float*)d_in[11]};
    const float* We[2] = {(const float*)d_in[6],  (const float*)d_in[12]};
    const float* aS[2] = {(const float*)d_in[7],  (const float*)d_in[13]};
    const float* aD[2] = {(const float*)d_in[8],  (const float*)d_in[14]};
    const float* aE[2] = {(const float*)d_in[9],  (const float*)d_in[15]};
    const float* bb[2] = {(const float*)d_in[10], (const float*)d_in[16]};

    const int N  = in_sizes[0] / CDIM;
    const int ED = in_sizes[6] / CDIM;      // We1 is [C, ED]
    const int E  = in_sizes[2] / ED;
    const int G  = out_size / CDIM;
    const long long* srcI = ei;
    const long long* dstI = ei + E;

    // ---- workspace carve (256B aligned regions) ----
    char* p = (char*)d_ws;
    auto carve = [&](size_t bytes) -> void* {
        void* r = (void*)p;
        p += (bytes + 255) & ~(size_t)255;
        return r;
    };
    float*    XL   = (float*)carve((size_t)N * CDIM * 4);   // x_l (GEMM out, both layers)
    float*    AGG  = (float*)carve((size_t)N * CDIM * 4);   // agg / h (reused both layers)
    bf16_t*   XH   = (bf16_t*)carve((size_t)N * CDIM * 2);
    bf16_t*   XLo  = (bf16_t*)carve((size_t)N * CDIM * 2);
    unsigned* BpkH = (unsigned*)carve(8192 * 4);
    unsigned* BpkL = (unsigned*)carve(8192 * 4);
    float*    vE   = (float*)carve(64 * 4);
    float*    oS   = (float*)carve((size_t)N * 4);
    float*    oD   = (float*)carve((size_t)N * 4);
    float*    cnt  = (float*)carve((size_t)N * 4);
    float*    sumT = (float*)carve((size_t)N * 4);
    unsigned* mE   = (unsigned*)carve((size_t)N * 4);
    float*    sA   = (float*)carve((size_t)N * 4);
    float*    pL   = (float*)carve((size_t)N * 4);
    float*    eB   = (float*)carve((size_t)E * 4);
    float*    gsum = (float*)carve((size_t)G * CDIM * 4);
    float*    gcnt = (float*)carve((size_t)G * 4);
    (void)ws_size; (void)n_in;

    const int TB = 256;
    const int NC = N * CDIM;

    hipMemsetAsync(cnt, 0, (size_t)N * 4, stream);
    k_count<<<(E + TB - 1) / TB, TB, 0, stream>>>(dstI, cnt, E);
    k_cvt_bf16<<<(NC + TB - 1) / TB, TB, 0, stream>>>(x, XH, XLo, NC);

    for (int L = 0; L < 2; ++L) {
        k_pack_w<<<(8192 + TB - 1) / TB, TB, 0, stream>>>(W[L], BpkH, BpkL);
        k_we_dot<<<1, 32, 0, stream>>>(We[L], aE[L], vE, ED);

        int tiles_m = (N + 15) / 16;
        int gthreads = tiles_m * 8 * 32;
        k_gemm_wmma<<<(gthreads + TB - 1) / TB, TB, 0, stream>>>(XH, XLo, BpkH, BpkL, XL, N);

        k_node_alpha<<<(N * 32 + TB - 1) / TB, TB, 0, stream>>>(XL, aS[L], aD[L], oS, oD, N);
        hipMemsetAsync(sumT, 0, (size_t)N * 4, stream);
        k_edge_t<<<(E + TB - 1) / TB, TB, 0, stream>>>(ea, vE, dstI, eB, sumT, E, ED);
        k_loop_init<<<(N + TB - 1) / TB, TB, 0, stream>>>(oS, oD, sumT, cnt, pL, mE, N);
        k_edge_max<<<(E + TB - 1) / TB, TB, 0, stream>>>(srcI, dstI, oS, oD, eB, mE, E);
        k_node_exp<<<(N + TB - 1) / TB, TB, 0, stream>>>(pL, mE, sA, N);
        k_edge_exp<<<(E + TB - 1) / TB, TB, 0, stream>>>(dstI, eB, mE, sA, E);
        k_self_agg<<<(N * 32 + TB - 1) / TB, TB, 0, stream>>>(XL, pL, sA, AGG, N);
        k_edge_agg<<<(E * 32 + TB - 1) / TB, TB, 0, stream>>>(srcI, dstI, XL, eB, sA, AGG, E);
        k_bias_silu<<<(NC + TB - 1) / TB, TB, 0, stream>>>(AGG, bb[L], NC);

        if (L == 0)  // h -> bf16 hi/lo for layer-2 GEMM
            k_cvt_bf16<<<(NC + TB - 1) / TB, TB, 0, stream>>>(AGG, XH, XLo, NC);
    }

    hipMemsetAsync(gsum, 0, (size_t)G * CDIM * 4, stream);
    hipMemsetAsync(gcnt, 0, (size_t)G * 4, stream);
    k_gpool<<<(N * 32 + TB - 1) / TB, TB, 0, stream>>>(AGG, bid, gsum, gcnt, N);
    k_gdiv<<<(G * CDIM + TB - 1) / TB, TB, 0, stream>>>(gsum, gcnt, (float*)d_out, G * CDIM);
}
